// PeriodBlock_61950608277567
// MI455X (gfx1250) — compile-verified, matrix operand order checked by reading, not run on hardware
//
#include <hip/hip_runtime.h>
#include <hip/hip_bf16.h>
#include <cstdint>
#include <cstddef>

typedef _Float16 half_t;
typedef __attribute__((ext_vector_type(16))) _Float16 v16h;
typedef __attribute__((ext_vector_type(8)))  _Float16 v8h;
typedef __attribute__((ext_vector_type(8)))  float    v8f;

#define EPI_GELU 1
#define EPI_ADD  2

__device__ __forceinline__ float gelu_exact(float y) {
  return 0.5f * y * (1.0f + erff(y * 0.70710678118654752f));
}

// Async global->LDS 16B move (CDNA5 TDM-lite path, tracked by ASYNCcnt).
// ldsOff = LDS byte offset (flat shared address truncated to 32 bits, per the
// ISA aperture rule); src = 16B-aligned global pointer.
__device__ __forceinline__ void async_ld_b128(unsigned ldsOff, const void* src) {
  asm volatile("global_load_async_to_lds_b128 %0, %1, off"
               :: "v"(ldsOff), "v"(src) : "memory");
}
__device__ __forceinline__ void wait_asynccnt0() {
  asm volatile("s_wait_asynccnt 0x0" ::: "memory");
}

// ---------------------------------------------------------------------------
// Generic f16 WMMA GEMM: C(M,Nact) = A(M,Kp) * Bw(Np,Kp)^T  (+bias, +epilogue)
// BM=64, BN=128, BK=32. 256 threads = 8 waves, each wave owns a 32x32 tile
// (2x2 v_wmma_f32_16x16x32_f16 accumulators, fp32 accumulation).
// Staging uses global_load_async_to_lds_b128 (no VGPR round trip).
// Row remap for conv-chunk scatter: orow = (m/wc)*pFull + j0 + (m%wc).
// ---------------------------------------------------------------------------
__global__ __launch_bounds__(256)
void k_gemm(const half_t* __restrict__ A, const half_t* __restrict__ Bw,
            const float* __restrict__ bias,
            float* __restrict__ Cf, half_t* __restrict__ Ch,
            int M, int Nact, int Kp, int ldC,
            int wc, int pFull, int j0, int flags)
{
  __shared__ half_t As[64 * 40];    // 64 rows x 32 halves, pad to 40
  __shared__ half_t Bs[128 * 40];   // 128 rows x 32 halves, pad to 40
  const int tid  = threadIdx.x;
  const int lane = tid & 31;
  const int wid  = tid >> 5;
  const int wm   = wid & 1;         // wave m-subtile (0..1)
  const int wn   = wid >> 1;        // wave n-subtile (0..3)
  const int lr   = lane & 15;
  const int sel  = lane >> 4;       // half-wave selector per ISA layout
  const int m0   = blockIdx.x * 64;
  const int n0   = blockIdx.y * 128;
  const int nk   = Kp >> 5;

  // Per-thread staging slots (uniform across iterations)
  const int rowA = tid >> 2, chA = tid & 3;
  const unsigned ldsA  = (unsigned)(uintptr_t)&As[rowA * 40 + chA * 8];
  const unsigned ldsB0 = (unsigned)(uintptr_t)&Bs[rowA * 40 + chA * 8];
  const unsigned ldsB1 = (unsigned)(uintptr_t)&Bs[(rowA + 64) * 40 + chA * 8];

  v8f c[2][2];
  c[0][0] = {}; c[0][1] = {}; c[1][0] = {}; c[1][1] = {};

  for (int kb = 0; kb < nk; ++kb) {
    if (kb) __syncthreads();        // everyone done reading previous tiles
    {
      const half_t* sA  = A  + (size_t)(m0 + rowA) * Kp + (kb << 5) + chA * 8;
      const half_t* sB0 = Bw + (size_t)(n0 + rowA) * Kp + (kb << 5) + chA * 8;
      const half_t* sB1 = Bw + (size_t)(n0 + 64 + rowA) * Kp + (kb << 5) + chA * 8;
      async_ld_b128(ldsA,  sA);     // LDS[...] <= MEM[...], ASYNCcnt++
      async_ld_b128(ldsB0, sB0);
      async_ld_b128(ldsB1, sB1);
    }
    if (kb + 1 < nk) {  // keep L2 ahead of the async engine
      __builtin_prefetch(A  + (size_t)(m0 + rowA) * Kp + ((kb + 1) << 5), 0, 0);
      __builtin_prefetch(Bw + (size_t)(n0 + rowA) * Kp + ((kb + 1) << 5), 0, 0);
    }
    wait_asynccnt0();               // our async writes landed in LDS
    __syncthreads();                // all waves' writes visible

    v16h a[2], b[2];
#pragma unroll
    for (int ms = 0; ms < 2; ++ms) {
      int r = wm * 32 + ms * 16 + lr;
      v8h lo = *(const v8h*)&As[r * 40 + sel * 8];
      v8h hi = *(const v8h*)&As[r * 40 + 16 + sel * 8];
#pragma unroll
      for (int q = 0; q < 8; ++q) { a[ms][q] = lo[q]; a[ms][q + 8] = hi[q]; }
    }
#pragma unroll
    for (int ns = 0; ns < 2; ++ns) {
      int r = wn * 32 + ns * 16 + lr;
      v8h lo = *(const v8h*)&Bs[r * 40 + sel * 8];
      v8h hi = *(const v8h*)&Bs[r * 40 + 16 + sel * 8];
#pragma unroll
      for (int q = 0; q < 8; ++q) { b[ns][q] = lo[q]; b[ns][q + 8] = hi[q]; }
    }
#pragma unroll
    for (int ms = 0; ms < 2; ++ms)
#pragma unroll
      for (int ns = 0; ns < 2; ++ns)
        c[ms][ns] = __builtin_amdgcn_wmma_f32_16x16x32_f16(
            false, a[ms], false, b[ns], (short)0, c[ms][ns], false, false);
  }

#pragma unroll
  for (int ms = 0; ms < 2; ++ms)
    for (int ns = 0; ns < 2; ++ns)
      for (int r = 0; r < 8; ++r) {
        int mrow = m0 + wm * 32 + ms * 16 + sel * 8 + r;  // f32 C/D layout
        int ncol = n0 + wn * 32 + ns * 16 + lr;
        if (ncol >= Nact) continue;
        float y = c[ms][ns][r] + (bias ? bias[ncol] : 0.0f);
        if (flags & EPI_GELU) y = gelu_exact(y);
        size_t orow = (size_t)(mrow / wc) * pFull + j0 + (mrow % wc);
        size_t oi = orow * (size_t)ldC + ncol;
        if (flags & EPI_ADD) Cf[oi] += y; else Cf[oi] = y;
        if (Ch) Ch[oi] = (half_t)y;
      }
}

// ---------------------------------------------------------------------------
// Build padded NHWC segment tensor (B,Pn,p,64) in f16 from x (B,720,64) f32.
// ---------------------------------------------------------------------------
__global__ void k_segment(const float* __restrict__ x, half_t* __restrict__ seg,
                          int Pn, int p, int padL)
{
  int m = blockIdx.x * blockDim.x + threadIdx.x;
  int total = 64 * Pn * p;
  if (m >= total) return;
  int j = m % p; int tmp = m / p; int i = tmp % Pn; int b = tmp / Pn;
  int t = i * p + j - padL;
  half_t* dst = seg + (size_t)m * 64;
  if (t >= 0 && t < 720) {
    const float* s = x + ((size_t)b * 720 + t) * 64;
    for (int c2 = 0; c2 < 64; ++c2) dst[c2] = (half_t)s[c2];
  } else {
    for (int c2 = 0; c2 < 64; ++c2) dst[c2] = (half_t)0.0f;
  }
}

// ---------------------------------------------------------------------------
// NHWC im2col for 3x3 SAME conv over width-chunk [j0, j0+wc). K = 9*64 = 576.
// One thread per (pixel, kh, kw): copies 64 contiguous channels (or zeros).
// ---------------------------------------------------------------------------
__global__ void k_im2col(const half_t* __restrict__ src, half_t* __restrict__ Aim,
                         int Pn, int p, int j0, int wc)
{
  size_t job = (size_t)blockIdx.x * blockDim.x + threadIdx.x;
  size_t M = (size_t)64 * Pn * wc;
  if (job >= M * 9) return;
  int kk = (int)(job % 9); size_t m = job / 9;
  int dh = kk / 3 - 1, dw = kk % 3 - 1;
  int jl = (int)(m % wc); size_t tmp = m / wc;
  int i = (int)(tmp % Pn); int b = (int)(tmp / Pn);
  int ii = i + dh, jj = jl + dw;
  half_t* dst = Aim + m * 576 + kk * 64;
  if (ii >= 0 && ii < Pn && jj >= 0 && jj < wc) {
    const half_t* s = src + (((size_t)b * Pn + ii) * p + j0 + jj) * 64;
    for (int c2 = 0; c2 < 64; c2 += 8) *(v8h*)(dst + c2) = *(const v8h*)(s + c2);
  } else {
    v8h z = {};
    for (int c2 = 0; c2 < 64; c2 += 8) *(v8h*)(dst + c2) = z;
  }
}

// Conv weight (64,64,3,3) f32 -> (128,576) f16, col = (kh*3+kw)*64 + cin.
__global__ void k_pack_convw(const float* __restrict__ w, half_t* __restrict__ Wp)
{
  int idx = blockIdx.x * blockDim.x + threadIdx.x;
  if (idx >= 128 * 576) return;
  int co = idx / 576, k = idx % 576;
  int kk = k / 64, ci = k % 64, dh = kk / 3, dw = kk % 3;
  float v = (co < 64) ? w[(((size_t)co * 64 + ci) * 3 + dh) * 3 + dw] : 0.0f;
  Wp[idx] = (half_t)v;
}

// Dense weight (Nact,Kact) f32 -> zero-padded (Np,Kp) f16.
__global__ void k_pack_dense(const float* __restrict__ src, half_t* __restrict__ dst,
                             int Nact, int Kact, int Np, int Kp)
{
  size_t idx = (size_t)blockIdx.x * blockDim.x + threadIdx.x;
  size_t total = (size_t)Np * Kp;
  if (idx >= total) return;
  int k = (int)(idx % Kp); int n = (int)(idx / Kp);
  float v = (n < Nact && k < Kact) ? src[(size_t)n * Kact + k] : 0.0f;
  dst[idx] = (half_t)v;
}

__global__ void k_fill_f32(float* p, size_t n)
{ size_t i = (size_t)blockIdx.x * blockDim.x + threadIdx.x; if (i < n) p[i] = 0.0f; }
__global__ void k_fill_f16(half_t* p, size_t n)
{ size_t i = (size_t)blockIdx.x * blockDim.x + threadIdx.x; if (i < n) p[i] = (half_t)0.0f; }

// ---------------------------------------------------------------------------
// out = out1 + out2 ; emit f32 canonical (B,Pn,p,64) and the three f16
// transposed MHA input layouts (mi1 K-padded to Kp1).
// ---------------------------------------------------------------------------
__global__ void k_combine(const float* __restrict__ o1, const float* __restrict__ o2,
                          float* __restrict__ outF,
                          half_t* __restrict__ mi1, half_t* __restrict__ mi2,
                          half_t* __restrict__ mi3,
                          int Pn, int p, int Kp1, int d2, int d3)
{
  size_t idx = (size_t)blockIdx.x * blockDim.x + threadIdx.x;
  size_t total = (size_t)64 * Pn * p * 64;
  if (idx >= total) return;
  int c = (int)(idx & 63); size_t m = idx >> 6;
  int j = (int)(m % p); size_t tmp = m / p;
  int i = (int)(tmp % Pn); int b = (int)(tmp / Pn);
  float v = o1[idx] + o2[idx];
  outF[idx] = v;
  half_t h = (half_t)v;
  mi1[((size_t)(b * 64 + c)) * Kp1 + i * p + j] = h;
  mi2[((size_t)(b * Pn + i)) * d2 + c * p + j] = h;
  mi3[((size_t)(b * p + j)) * d3 + c * Pn + i] = h;
}

// ---------------------------------------------------------------------------
// Attention core: one block per (b,si) row; loops heads. Tiny FLOP share,
// f32 math, writes o as f16 into the K-padded out-projection input.
// ---------------------------------------------------------------------------
__global__ __launch_bounds__(128)
void k_attn(const float* __restrict__ qkv, half_t* __restrict__ oh,
            int S, int d, int nh, int Kp)
{
  int bs = blockIdx.x;          // b*S + si
  int b  = bs / S;
  int dh = d / nh;
  float scale = rsqrtf((float)dh);
  __shared__ float lg[64];
  for (int h = 0; h < nh; ++h) {
    const float* q = qkv + (size_t)bs * 3 * d + h * dh;
    if ((int)threadIdx.x < S) {
      int sj = threadIdx.x;
      const float* kr = qkv + ((size_t)(b * S + sj)) * 3 * d + d + h * dh;
      float acc = 0.0f;
      for (int k2 = 0; k2 < dh; ++k2) acc += q[k2] * kr[k2];
      lg[sj] = acc * scale;
    }
    __syncthreads();
    if (threadIdx.x == 0) {
      float mx = -1e30f;
      for (int j = 0; j < S; ++j) mx = fmaxf(mx, lg[j]);
      float s = 0.0f;
      for (int j = 0; j < S; ++j) { lg[j] = __expf(lg[j] - mx); s += lg[j]; }
      float inv = 1.0f / s;
      for (int j = 0; j < S; ++j) lg[j] *= inv;
    }
    __syncthreads();
    for (int k2 = threadIdx.x; k2 < dh; k2 += blockDim.x) {
      float acc = 0.0f;
      const float* vb = qkv + (size_t)(b * S) * 3 * d + 2 * d + h * dh + k2;
      for (int sj = 0; sj < S; ++sj) acc += lg[sj] * vb[(size_t)sj * 3 * d];
      oh[(size_t)bs * Kp + h * dh + k2] = (half_t)acc;
    }
    __syncthreads();
  }
}

// ---------------------------------------------------------------------------
// Single-bin DFT amplitude (bins 720/24=30 and 720/12=60), mean over N,
// softmax over the two periods. Replaces the full rFFT.
// ---------------------------------------------------------------------------
__global__ void k_influence(const float* __restrict__ x, float* __restrict__ infl)
{
  int b = blockIdx.x;      // 64 blocks, 64 threads (=channels)
  int c = threadIdx.x;
  __shared__ float amp[2][64];
  for (int pi = 0; pi < 2; ++pi) {
    int f = (pi == 0) ? 30 : 60;
    float w = -2.0f * 3.14159265358979323846f * (float)f / 720.0f;
    float re = 0.0f, im = 0.0f;
    for (int t = 0; t < 720; ++t) {
      float v = x[((size_t)b * 720 + t) * 64 + c];
      float sn, cs; sincosf(w * (float)t, &sn, &cs);
      re += v * cs; im += v * sn;
    }
    amp[pi][c] = sqrtf(re * re + im * im);
  }
  __syncthreads();
  if (c == 0) {
    float a0 = 0.0f, a1 = 0.0f;
    for (int i = 0; i < 64; ++i) { a0 += amp[0][i]; a1 += amp[1][i]; }
    a0 *= (1.0f / 64.0f); a1 *= (1.0f / 64.0f);
    float mx = fmaxf(a0, a1);
    float e0 = __expf(a0 - mx), e1 = __expf(a1 - mx);
    float inv = 1.0f / (e0 + e1);
    infl[b * 2 + 0] = e0 * inv;
    infl[b * 2 + 1] = e1 * inv;
  }
}

// ---------------------------------------------------------------------------
// Fused LambdaMixer softmax + branch gather + unpad + influence mix + residual.
// ---------------------------------------------------------------------------
__global__ void k_final(const float* __restrict__ x, const float* __restrict__ infl,
                        const float* __restrict__ mlog,
                        const float* o0, const float* a0, const float* b0, const float* c0,
                        const float* o1, const float* a1, const float* b1, const float* c1,
                        float* __restrict__ res)
{
  int idx = blockIdx.x * blockDim.x + threadIdx.x;
  if (idx >= 64 * 720 * 64) return;
  int c = idx & 63;
  int t = (idx >> 6) % 720;
  int b = idx / (720 * 64);
  float l0 = mlog[c], l1 = mlog[64 + c], l2 = mlog[128 + c], l3 = mlog[192 + c];
  float mx = fmaxf(fmaxf(l0, l1), fmaxf(l2, l3));
  float e0 = __expf(l0 - mx), e1 = __expf(l1 - mx), e2 = __expf(l2 - mx), e3 = __expf(l3 - mx);
  float inv = 1.0f / (e0 + e1 + e2 + e3);
  float lam0 = e0 * inv, lam1 = e1 * inv, lam2 = e2 * inv, lam3 = e3 * inv;
  float acc = x[idx];
  {   // period 0: p=24, Pn=31, padL=12, d1=744, d2=1536, d3=1984
    int tt = t + 12; int i = tt / 24, j = tt - i * 24;
    float o  = o0[((size_t)(b * 31 + i) * 24 + j) * 64 + c];
    float v1 = a0[(size_t)(b * 64 + c) * 744 + i * 24 + j];
    float v2 = b0[(size_t)(b * 31 + i) * 1536 + c * 24 + j];
    float v3 = c0[(size_t)(b * 24 + j) * 1984 + c * 31 + i];
    acc += infl[b * 2 + 0] * (lam0 * o + lam1 * v1 + lam2 * v2 + lam3 * v3);
  }
  {   // period 1: p=12, Pn=61, padL=6, d1=732, d2=768, d3=3904
    int tt = t + 6; int i = tt / 12, j = tt - i * 12;
    float o  = o1[((size_t)(b * 61 + i) * 12 + j) * 64 + c];
    float v1 = a1[(size_t)(b * 64 + c) * 732 + i * 12 + j];
    float v2 = b1[(size_t)(b * 61 + i) * 768 + c * 12 + j];
    float v3 = c1[(size_t)(b * 12 + j) * 3904 + c * 61 + i];
    acc += infl[b * 2 + 1] * (lam0 * o + lam1 * v1 + lam2 * v2 + lam3 * v3);
  }
  res[idx] = acc;
}

// ---------------------------------------------------------------------------
// Host orchestration
// ---------------------------------------------------------------------------
extern "C" void kernel_launch(void* const* d_in, const int* in_sizes, int n_in,
                              void* d_out, int out_size, void* d_ws, size_t ws_size,
                              hipStream_t stream)
{
  (void)in_sizes; (void)n_in; (void)out_size; (void)ws_size;
  const float* x     = (const float*)d_in[0];
  const float* mixer = (const float*)d_in[50];

  char* base = (char*)d_ws;
  size_t off = 0;
  auto alloc = [&](size_t bytes) -> char* {
    char* p = base + off;
    off = (off + bytes + 255) & ~(size_t)255;
    return p;
  };

  // Shared (sequentially reused) scratch
  half_t* Aim   = (half_t*)alloc((size_t)47616 * 576 * 2);   // max im2col A
  half_t* Wpack = (half_t*)alloc((size_t)11776 * 3904 * 2);  // max packed weight
  float*  qkvF  = (float*) alloc((size_t)9142272 * 4);       // max (B*S)x(3d)
  half_t* oH    = (half_t*)alloc((size_t)3145728 * 2);       // max (B*S)xKp
  float*  infl  = (float*) alloc(128 * 4);

  struct PD { int p, Pn, padL, d1, Kp1, d2, d3; };
  const PD per[2] = { {24, 31, 12, 744, 768, 1536, 1984},
                      {12, 61,  6, 732, 736,  768, 3904} };

  half_t *seg[2], *out1H[2], *mi1[2], *mi2[2], *mi3[2];
  float  *out1F[2], *out2F[2], *outF[2], *m1F[2], *m2F[2], *m3F[2];
  for (int pe = 0; pe < 2; ++pe) {
    const PD& pp = per[pe];
    size_t M = (size_t)64 * pp.Pn * pp.p;
    seg[pe]   = (half_t*)alloc(M * 64 * 2);
    out1F[pe] = (float*) alloc(M * 64 * 4);
    out1H[pe] = (half_t*)alloc(M * 64 * 2);
    out2F[pe] = (float*) alloc(M * 64 * 4);
    outF[pe]  = (float*) alloc(M * 64 * 4);
    mi1[pe]   = (half_t*)alloc((size_t)4096 * pp.Kp1 * 2);
    mi2[pe]   = (half_t*)alloc((size_t)64 * pp.Pn * pp.d2 * 2);
    mi3[pe]   = (half_t*)alloc((size_t)64 * pp.p  * pp.d3 * 2);
    m1F[pe]   = (float*) alloc((size_t)4096 * pp.d1 * 4);
    m2F[pe]   = (float*) alloc((size_t)64 * pp.Pn * pp.d2 * 4);
    m3F[pe]   = (float*) alloc((size_t)64 * pp.p  * pp.d3 * 4);
  }

  auto gemm = [&](const half_t* A, const half_t* Bw, const float* bias,
                  float* Cf, half_t* Ch, int M, int Nact, int Kp, int ldC,
                  int wcv, int pF, int j0v, int flags) {
    dim3 g(M / 64, (Nact + 127) / 128);
    k_gemm<<<g, 256, 0, stream>>>(A, Bw, bias, Cf, Ch, M, Nact, Kp, ldC,
                                  wcv, pF, j0v, flags);
  };
  auto packW = [&](const float* src, int Nact, int Kact, int Np, int Kp) {
    size_t tot = (size_t)Np * Kp;
    k_pack_dense<<<(unsigned)((tot + 255) / 256), 256, 0, stream>>>(src, Wpack, Nact, Kact, Np, Kp);
  };

  for (int pe = 0; pe < 2; ++pe) {
    const PD& pp = per[pe];
    int M = 64 * pp.Pn * pp.p;

    // 1) padded segment (NHWC, f16)
    k_segment<<<(M + 255) / 256, 256, 0, stream>>>(x, seg[pe], pp.Pn, pp.p, pp.padL);

    // 2) conv2: implicit GEMM (K=576) + bias + exact GELU; dual f32/f16 store
    const float* w2 = (const float*)d_in[2 + pe * 2];
    const float* b2 = (const float*)d_in[3 + pe * 2];
    k_pack_convw<<<(128 * 576) / 256, 256, 0, stream>>>(w2, Wpack);
    k_im2col<<<(unsigned)(((size_t)M * 9 + 255) / 256), 256, 0, stream>>>(
        seg[pe], Aim, pp.Pn, pp.p, 0, pp.p);
    gemm(Aim, Wpack, b2, out1F[pe], out1H[pe], M, 64, 576, 64,
         pp.p, pp.p, 0, EPI_GELU);

    // 3) conv3 part-chunks, accumulated into out2
    k_fill_f32<<<(unsigned)(((size_t)M * 64 + 255) / 256), 256, 0, stream>>>(
        out2F[pe], (size_t)M * 64);
    int cbase = 6 + pe * 10;
    const int partsList[2] = {2, 3};
    int wIdx = 0;
    for (int lv = 0; lv < 2; ++lv) {
      int parts = partsList[lv];
      int wcv = pp.p / parts;
      int Mc = 64 * pp.Pn * wcv;
      for (int pt = 0; pt < parts; ++pt) {
        const float* w3 = (const float*)d_in[cbase + wIdx * 2];
        const float* b3 = (const float*)d_in[cbase + wIdx * 2 + 1];
        ++wIdx;
        k_pack_convw<<<(128 * 576) / 256, 256, 0, stream>>>(w3, Wpack);
        k_im2col<<<(unsigned)(((size_t)Mc * 9 + 255) / 256), 256, 0, stream>>>(
            out1H[pe], Aim, pp.Pn, pp.p, pt * wcv, wcv);
        gemm(Aim, Wpack, b3, out2F[pe], nullptr, Mc, 64, 576, 64,
             wcv, pp.p, pt * wcv, EPI_GELU | EPI_ADD);
      }
    }

    // 4) combine + emit the three MHA input layouts (mi1 zero-padded in K)
    k_fill_f16<<<(unsigned)(((size_t)4096 * pp.Kp1 + 255) / 256), 256, 0, stream>>>(
        mi1[pe], (size_t)4096 * pp.Kp1);
    k_combine<<<(unsigned)(((size_t)M * 64 + 255) / 256), 256, 0, stream>>>(
        out1F[pe], out2F[pe], outF[pe], mi1[pe], mi2[pe], mi3[pe],
        pp.Pn, pp.p, pp.Kp1, pp.d2, pp.d3);

    // 5) the three MHAs
    const int dd[3]  = { pp.d1, pp.d2, pp.d3 };
    const int Kpd[3] = { pp.Kp1, pp.d2, pp.d3 };
    const int Ss[3]  = { 64, pp.Pn, pp.p };
    const int nhs[3] = { 6, 4, 4 };
    const half_t* Ain[3] = { mi1[pe], mi2[pe], mi3[pe] };
    float* mOut[3] = { m1F[pe], m2F[pe], m3F[pe] };
    for (int l = 0; l < 3; ++l) {
      int mb = 26 + pe * 12 + l * 4;
      const float* wqkv = (const float*)d_in[mb];
      const float* bq   = (const float*)d_in[mb + 1];
      const float* wo   = (const float*)d_in[mb + 2];
      const float* bo   = (const float*)d_in[mb + 3];
      int d = dd[l], Kp = Kpd[l], S = Ss[l], Mr = 64 * S;
      int N3 = 3 * d, Np3 = ((N3 + 127) / 128) * 128;
      packW(wqkv, N3, d, Np3, Kp);
      gemm(Ain[l], Wpack, bq, qkvF, nullptr, Mr, N3, Kp, N3, 1, 1, 0, 0);
      k_fill_f16<<<(unsigned)(((size_t)Mr * Kp + 255) / 256), 256, 0, stream>>>(
          oH, (size_t)Mr * Kp);
      k_attn<<<Mr, 128, 0, stream>>>(qkvF, oH, S, d, nhs[l], Kp);
      int NpO = ((d + 127) / 128) * 128;
      packW(wo, d, d, NpO, Kp);
      gemm(oH, Wpack, bo, mOut[l], nullptr, Mr, d, Kp, d, 1, 1, 0, 0);
    }
  }

  // 6) period influence + fused mixer/unpad/residual
  k_influence<<<64, 64, 0, stream>>>(x, infl);
  k_final<<<(64 * 720 * 64 + 255) / 256, 256, 0, stream>>>(
      x, infl, mixer,
      outF[0], m1F[0], m2F[0], m3F[0],
      outF[1], m1F[1], m2F[1], m3F[1],
      (float*)d_out);
}
